// PtConv_37297495998476
// MI455X (gfx1250) — matrix operand order repaired and checked by reading, not compiled, not run on hardware
//
#include <hip/hip_runtime.h>

// PtConv (KPConv-style) fused kernel for gfx1250 (MI455X).
// ALL matrix work (per-pair MLP, per-point aggregation, final conv GEMM)
// on v_wmma_f32_16x16x32_f16 (f16 in, f32 accumulate).

typedef __attribute__((ext_vector_type(16))) _Float16 v16h;
typedef __attribute__((ext_vector_type(8)))  float    v8f;

union V16H { v16h v; _Float16 h[16]; };
union V8F  { v8f  v; float    f[8];  };

#define BB   8
#define NN   8192
#define CIN  64
#define COUT 128
#define KNB  16
#define MM   16
#define CM   1024   // CIN*MM

// A fragment (16x32, f16): base = &arr[row*stride + kchunk*32 + 16*hi]
// halves 0..15 <- base[0..15] (K = kk*32 + 16hi + h)
__device__ __forceinline__ v16h load_a_frag(const _Float16* base) {
    V16H a;
    #pragma unroll
    for (int h = 0; h < 8; ++h) { a.h[h] = base[h]; a.h[8 + h] = base[8 + h]; }
    return a.v;
}

// B fragment (32x16, f16): base = &arr[col*stride + kchunk*32 + 16*hi]
// halves 0..15 <- base[0..15] (K = kk*32 + 16hi + h)
__device__ __forceinline__ v16h load_b_frag(const _Float16* base) {
    V16H b;
    #pragma unroll
    for (int h = 0; h < 16; ++h) b.h[h] = base[h];
    return b.v;
}

// ---------- prep kernels ----------

// WT[co][c*16+m] = (f16) weight[c][m][co]   (weight is [CIN, M, COUT] row-major)
__global__ void cvt_weight_kernel(const float* __restrict__ w, _Float16* __restrict__ wt) {
    int i = blockIdx.x * 256 + threadIdx.x;        // i over COUT*CM
    if (i >= COUT * CM) return;
    int co = i >> 10;
    int cm = i & (CM - 1);
    wt[i] = (_Float16)w[(size_t)cm * COUT + co];
}

__global__ void cvt_feat_kernel(const float* __restrict__ f, _Float16* __restrict__ fh, int n) {
    int i = blockIdx.x * 256 + threadIdx.x;
    if (i < n) fh[i] = (_Float16)f[i];
}

__global__ void copy_pts_kernel(const float* __restrict__ src, float* __restrict__ dst, int n) {
    int i = blockIdx.x * 256 + threadIdx.x;
    if (i < n) dst[i] = src[i];
}

// ---------- fused main kernel ----------
// grid: 4096 workgroups (B * N/16), block: 128 threads (4 wave32)

__global__ __launch_bounds__(128)
void ptconv_main_kernel(const float* __restrict__ input_pts,
                        const float* __restrict__ output_pts,
                        const int*   __restrict__ indices,
                        const float* __restrict__ centers,
                        const float* __restrict__ bias,
                        const float* __restrict__ l1w, const float* __restrict__ l1b,
                        const float* __restrict__ l2w, const float* __restrict__ l2b,
                        const float* __restrict__ l3w, const float* __restrict__ l3b,
                        const _Float16* __restrict__ Ff,   // features f16 [B][N][CIN]
                        const _Float16* __restrict__ WT,   // weight  f16 [COUT][CM]
                        float* __restrict__ out) {
    __shared__ float sL1b[32];
    __shared__ float sL2b[16];
    __shared__ float sL3b[16];
    __shared__ float sCen[48];        // [dim][M]
    __shared__ float sOutP[16 * 3];   // 16 tile points
    __shared__ __align__(16) _Float16 sW1h[32 * 64];       // l1_w [32][48] K-pad->64
    __shared__ __align__(16) _Float16 sW2h[16 * 32];       // l2_w [16][32]
    __shared__ __align__(16) _Float16 sW3h[16 * 32];       // l3_w [16][16] K-pad->32
    __shared__ __align__(16) _Float16 sIn[256 * 64];       // MLP input rows, K-pad->64
    __shared__ __align__(16) _Float16 sH1[256 * 32];       // layer1 out
    __shared__ __align__(16) _Float16 sH2[256 * 32];       // layer2 out, K-pad->32
    __shared__ __align__(16) _Float16 sDm[16 * 16 * 16];   // [p][m][k]   D^T
    __shared__ __align__(16) _Float16 sFt[16 * 64 * 16];   // [p][c][k]   gathered feats
    __shared__ __align__(16) _Float16 sX [16 * 1024];      // [p][c*16+m] flattened agg
    __shared__ __align__(16) _Float16 sZero[64];           // branchless K-pad source

    const int tid  = threadIdx.x;
    const int g    = blockIdx.x;
    const int b    = g >> 9;                 // 512 tiles per batch
    const int n0   = (g & 511) << 4;         // first of 16 points in tile
    const int lane = tid & 31;
    const int wave = tid >> 5;
    const int lo   = lane & 15;
    const int hi   = lane >> 4;              // 0 or 1

    // ---- phase 0: stage MLP params (f16, zero-padded K) + tile points ----
    for (int i = tid; i < 32 * 64; i += 128) {
        int co = i >> 6, j = i & 63;
        sW1h[i] = (_Float16)(j < 48 ? l1w[co * 48 + j] : 0.0f);
    }
    for (int i = tid; i < 16 * 32; i += 128) sW2h[i] = (_Float16)l2w[i];
    for (int i = tid; i < 16 * 32; i += 128) {
        int m = i >> 5, j = i & 31;
        sW3h[i] = (_Float16)(j < 16 ? l3w[m * 16 + j] : 0.0f);
    }
    if (tid < 64) sZero[tid] = (_Float16)0.0f;
    if (tid < 32) sL1b[tid] = l1b[tid];
    if (tid < 16) { sL2b[tid] = l2b[tid]; sL3b[tid] = l3b[tid]; }
    if (tid < 48) {
        sCen[tid]  = centers[tid];
        sOutP[tid] = output_pts[((size_t)b * NN + n0) * 3 + tid];
    }
    __syncthreads();

    // ---- phase 1a: build MLP input rows + gather neighbor features ----
    for (int t = tid; t < 256; t += 128) {
        const int p = t >> 4;
        const int k = t & 15;
        const int idx = indices[((size_t)b * NN + n0 + p) * KNB + k];

        const float* ip = input_pts + ((size_t)b * NN + idx) * 3;
        const float rel0 = ip[0] - sOutP[p * 3 + 0];
        const float rel1 = ip[1] - sOutP[p * 3 + 1];
        const float rel2 = ip[2] - sOutP[p * 3 + 2];

        _Float16* rowp = &sIn[t * 64];
        #pragma unroll
        for (int mi = 0; mi < 16; ++mi) {
            rowp[mi]      = (_Float16)(rel0 - sCen[mi]);
            rowp[16 + mi] = (_Float16)(rel1 - sCen[16 + mi]);
            rowp[32 + mi] = (_Float16)(rel2 - sCen[32 + mi]);
            rowp[48 + mi] = (_Float16)0.0f;                   // K pad 48->64
        }

        // gather 64 f16 neighbor features, store transposed [p][c][k]
        const _Float16* fs = Ff + ((size_t)b * NN + idx) * CIN;
        _Float16 fl[64];
        #pragma unroll
        for (int c = 0; c < 64; ++c) fl[c] = fs[c];
        #pragma unroll
        for (int c = 0; c < 64; ++c) sFt[(p * 64 + c) * 16 + k] = fl[c];
    }
    __syncthreads();

    // ---- phase 1b: MLP layer1 [256x48(->64)] x [48x32] + bias + relu ----
    for (int i = 0; i < 4; ++i) {
        const int mt  = wave * 4 + i;           // M-tile of 16 rows
        const int row = mt * 16 + lo;
        #pragma unroll
        for (int nt = 0; nt < 2; ++nt) {
            V8F acc;
            #pragma unroll
            for (int r = 0; r < 8; ++r) acc.f[r] = 0.0f;
            #pragma unroll
            for (int kk = 0; kk < 2; ++kk) {
                v16h a  = load_a_frag(&sIn[row * 64 + kk * 32 + hi * 16]);
                v16h bm = load_b_frag(&sW1h[(nt * 16 + lo) * 64 + kk * 32 + hi * 16]);
                acc.v = __builtin_amdgcn_wmma_f32_16x16x32_f16(
                    false, a, false, bm, (short)0, acc.v, false, false);
            }
            #pragma unroll
            for (int r = 0; r < 8; ++r) {
                const int rr  = mt * 16 + r + hi * 8;
                const int col = nt * 16 + lo;
                sH1[rr * 32 + col] = (_Float16)fmaxf(acc.f[r] + sL1b[col], 0.0f);
            }
        }
    }
    __syncthreads();

    // ---- phase 1c: MLP layer2 [256x32] x [32x16] + bias + relu ----
    for (int i = 0; i < 4; ++i) {
        const int mt  = wave * 4 + i;
        const int row = mt * 16 + lo;
        v16h a  = load_a_frag(&sH1[row * 32 + hi * 16]);
        v16h bm = load_b_frag(&sW2h[lo * 32 + hi * 16]);
        V8F acc;
        #pragma unroll
        for (int r = 0; r < 8; ++r) acc.f[r] = 0.0f;
        acc.v = __builtin_amdgcn_wmma_f32_16x16x32_f16(
            false, a, false, bm, (short)0, acc.v, false, false);
        #pragma unroll
        for (int r = 0; r < 8; ++r) {
            const int rr = mt * 16 + r + hi * 8;
            sH2[rr * 32 + lo]      = (_Float16)fmaxf(acc.f[r] + sL2b[lo], 0.0f);
            sH2[rr * 32 + 16 + lo] = (_Float16)0.0f;          // K pad 16->32
        }
    }
    __syncthreads();

    // ---- phase 1d: MLP layer3 [256x16(->32)] x [16x16] + bias + relu -> D^T ----
    for (int i = 0; i < 4; ++i) {
        const int mt  = wave * 4 + i;           // mt == point p for these rows
        const int row = mt * 16 + lo;
        v16h a  = load_a_frag(&sH2[row * 32 + hi * 16]);
        v16h bm = load_b_frag(&sW3h[lo * 32 + hi * 16]);
        V8F acc;
        #pragma unroll
        for (int r = 0; r < 8; ++r) acc.f[r] = 0.0f;
        acc.v = __builtin_amdgcn_wmma_f32_16x16x32_f16(
            false, a, false, bm, (short)0, acc.v, false, false);
        // row = mt*16 + r + 8hi -> p = mt, k = r + 8hi ; col m = lo
        // sDm[(p*16+m)*16 + k]: 8 consecutive halves per lane -> packed store
        #pragma unroll
        for (int r = 0; r < 8; ++r) {
            sDm[(mt * 16 + lo) * 16 + hi * 8 + r] =
                (_Float16)fmaxf(acc.f[r] + sL3b[lo], 0.0f);
        }
    }
    __syncthreads();

    // ---- phase 2: per-point aggregation aggT(16x64) = D^T(16x16,pad K->32) x F(16x64) ----
    for (int pp = 0; pp < 4; ++pp) {
        const int p = wave * 4 + pp;

        // A fragment: a[h] = Dt[m=lo][k = h] for hi=0; hi=1 -> K>=16 pad = 0
        V16H a;
        {
            const _Float16* dp = &sDm[(p * 16 + lo) * 16];
            #pragma unroll
            for (int h = 0; h < 8; ++h) { a.h[h] = hi ? (_Float16)0.0f : dp[h]; a.h[8 + h] = hi ? (_Float16)0.0f : dp[8 + h]; }
        }

        #pragma unroll
        for (int ct = 0; ct < 4; ++ct) {
            // B fragment: K = h + 16*hi; rows K>=16 zero-pad via branchless pointer select
            const _Float16* fp = hi ? sZero : &sFt[(p * 64 + ct * 16 + lo) * 16];
            v16h bm = load_b_frag(fp);

            V8F acc;
            #pragma unroll
            for (int r = 0; r < 8; ++r) acc.f[r] = 0.0f;
            acc.v = __builtin_amdgcn_wmma_f32_16x16x32_f16(
                false, a.v, false, bm, (short)0, acc.v, false, false);

            // acc[r] = aggT[m = r + 8*hi][ct*16 + lo]  ->  X[p][c*16 + m]
            #pragma unroll
            for (int r = 0; r < 8; ++r) {
                const int m = r + hi * 8;
                const int c = ct * 16 + lo;
                sX[p * 1024 + c * 16 + m] = (_Float16)acc.f[r];
            }
        }
    }
    __syncthreads();

    // ---- phase 3: out(16x128) = X(16x1024) x W(1024x128), WMMA K-loop ----
    const float inv_k = 1.0f / (float)KNB;
    for (int nti = 0; nti < 2; ++nti) {
        const int nt = wave + nti * 4;        // N-tile: cols [nt*16, nt*16+16)

        V8F acc;
        #pragma unroll
        for (int r = 0; r < 8; ++r) acc.f[r] = 0.0f;

        for (int kk = 0; kk < 32; ++kk) {
            v16h a  = load_a_frag(&sX[lo * 1024 + kk * 32 + hi * 16]);
            v16h bm = load_b_frag(WT + (size_t)(nt * 16 + lo) * CM + kk * 32 + hi * 16);
            acc.v = __builtin_amdgcn_wmma_f32_16x16x32_f16(
                false, a, false, bm, (short)0, acc.v, false, false);
        }

        #pragma unroll
        for (int r = 0; r < 8; ++r) {
            const int m  = r + hi * 8;
            const int co = nt * 16 + lo;
            out[((size_t)b * NN + n0 + m) * COUT + co] = acc.f[r] * inv_k + bias[co];
        }
    }
}

// ---------- launcher ----------

extern "C" void kernel_launch(void* const* d_in, const int* in_sizes, int n_in,
                              void* d_out, int out_size, void* d_ws, size_t ws_size,
                              hipStream_t stream) {
    const float* features   = (const float*)d_in[0];
    const float* input_pts  = (const float*)d_in[1];
    const float* output_pts = (const float*)d_in[2];
    const int*   indices    = (const int*)  d_in[3];
    const float* centers    = (const float*)d_in[4];
    const float* weight     = (const float*)d_in[5];
    const float* bias       = (const float*)d_in[6];
    const float* l1w        = (const float*)d_in[7];
    const float* l1b        = (const float*)d_in[8];
    const float* l2w        = (const float*)d_in[9];
    const float* l2b        = (const float*)d_in[10];
    const float* l3w        = (const float*)d_in[11];
    const float* l3b        = (const float*)d_in[12];

    _Float16* WT = (_Float16*)d_ws;                                   // 256 KB
    _Float16* Ff = (_Float16*)((char*)d_ws + (size_t)COUT * CM * 2);  // 8 MB

    float* out     = (float*)d_out;
    float* pts_dst = out + (size_t)BB * NN * COUT;   // second tuple output

    const int nw = COUT * CM;          // 131072
    const int nf = BB * NN * CIN;      // 4194304
    const int np = BB * NN * 3;        // 196608

    cvt_weight_kernel<<<(nw + 255) / 256, 256, 0, stream>>>(weight, WT);
    cvt_feat_kernel  <<<(nf + 255) / 256, 256, 0, stream>>>(features, Ff, nf);
    copy_pts_kernel  <<<(np + 255) / 256, 256, 0, stream>>>(output_pts, pts_dst, np);

    ptconv_main_kernel<<<dim3(BB * NN / 16), dim3(128), 0, stream>>>(
        input_pts, output_pts, indices, centers, bias,
        l1w, l1b, l2w, l2b, l3w, l3b, Ff, WT, out);
}
